// WindowCrossAttention_78554951844114
// MI455X (gfx1250) — compile-verified, matrix operand order checked by reading, not run on hardware
//
#include <hip/hip_runtime.h>
#include <hip/hip_bf16.h>

// ---------------------------------------------------------------------------
// Window cross attention (Swin-like), MI455X / gfx1250, wave32 + WMMA bf16.
// B=4096 windows, N=64 tokens, C=128, H=4 heads, HD=32, nW(mask groups)=1024.
// One workgroup (8 waves) per window; all intermediates in LDS.
// ---------------------------------------------------------------------------

typedef __attribute__((ext_vector_type(16))) __bf16 v16bf;
typedef __attribute__((ext_vector_type(8)))  __bf16 v8bf;
typedef __attribute__((ext_vector_type(8)))  float  v8f;

#define NTOK   64
#define CCH    128
#define QKVC   384           // 3*C
#define NHEAD  4
#define HD     32
#define NWIN   4096          // B
#define MASKW  1023          // nW-1

// workspace layout (bytes)
#define WS_WQKVT   0         // 2 * 384*128 bf16  = 196608 B
#define WS_WPROJT  196608    // 2 * 128*128 bf16  =  65536 B
#define WS_BIAS    262144    // 2 * 4*64*64 f32   = 131072 B

// LDS layout (bytes), total 295936 (<= 320KB/WGP)
#define LDS_X      0         // 64*128 f32 = 32768
#define LDS_Y      32768     // 32768
#define LDS_QKVX   65536     // 64*384 bf16 = 49152
#define LDS_QKVY   114688    // 49152
#define LDS_VT     163840    // 2*128*64 bf16 = 32768 (V transposed: [s][d][m])
#define LDS_P      196608    // 8 waves * 64*64 bf16 = 65536 ; union with ProjBuf f32
#define LDS_OBUF   262144    // 2*64*128 bf16 = 32768
#define LDS_STATS  294912    // 128 mu + 128 rstd f32 = 1024
#define SMEM_BYTES 295936

__device__ __forceinline__ v8f wmma_bf16(v16bf a, v16bf b, v8f c) {
  return __builtin_amdgcn_wmma_f32_16x16x32_bf16(false, a, false, b, (short)0, c,
                                                 false, false);
}

// 16x32 bf16 fragment, per-lane-contiguous K layout (A-matrix layout; also the
// B-matrix layout when the source is stored transposed, i.e. [col][k]).
// lane 0..15 -> index tile*16+(lane&15), K chunk 0..7 & 16..23
// lane 16..31 -> same index set, K chunk 8..15 & 24..31
__device__ __forceinline__ v16bf frag_bf(const __bf16* base, int stride,
                                         int tile, int kk, int lane) {
  int r  = tile * 16 + (lane & 15);
  int kb = kk + ((lane & 16) ? 8 : 0);
  const __bf16* p = base + r * stride + kb;
  v8bf lo = *reinterpret_cast<const v8bf*>(p);
  v8bf hi = *reinterpret_cast<const v8bf*>(p + 16);
  return __builtin_shufflevector(lo, hi, 0, 1, 2, 3, 4, 5, 6, 7,
                                          8, 9, 10, 11, 12, 13, 14, 15);
}

__device__ __forceinline__ v16bf frag_f32(const float* base, int stride,
                                          int tile, int kk, int lane) {
  int r  = tile * 16 + (lane & 15);
  int kb = kk + ((lane & 16) ? 8 : 0);
  const float* p = base + r * stride + kb;
  v16bf a;
#pragma unroll
  for (int i = 0; i < 8; ++i) {
    a[i]     = (__bf16)p[i];
    a[i + 8] = (__bf16)p[16 + i];
  }
  return a;
}

// --------------------------- prep kernels ----------------------------------

// wqkv (C,3C) -> bf16 transposed [s][j][k], j in [0,384), k in [0,128)
__global__ void prep_wqkv(const float* __restrict__ w1,
                          const float* __restrict__ w2,
                          __bf16* __restrict__ out) {
  int idx = blockIdx.x * 256 + threadIdx.x;
  if (idx >= 2 * QKVC * CCH) return;
  int s = idx / (QKVC * CCH);
  int rem = idx % (QKVC * CCH);
  int j = rem / CCH, k = rem % CCH;
  const float* w = s ? w2 : w1;
  out[idx] = (__bf16)w[k * QKVC + j];
}

// wproj (C,C) -> bf16 transposed [s][j][k]
__global__ void prep_wproj(const float* __restrict__ w1,
                           const float* __restrict__ w2,
                           __bf16* __restrict__ out) {
  int idx = blockIdx.x * 256 + threadIdx.x;
  if (idx >= 2 * CCH * CCH) return;
  int s = idx / (CCH * CCH);
  int rem = idx % (CCH * CCH);
  int j = rem / CCH, k = rem % CCH;
  const float* w = s ? w2 : w1;
  out[idx] = (__bf16)w[k * CCH + j];
}

// bias[s][h][n][m] = tbl_s[rpi[n][m]][h]
__global__ void prep_bias(const float* __restrict__ tbl1,
                          const float* __restrict__ tbl2,
                          const int* __restrict__ rpi,
                          float* __restrict__ bias) {
  int idx = blockIdx.x * 256 + threadIdx.x;   // 2*64*64
  if (idx >= 2 * NTOK * NTOK) return;
  int s = idx / (NTOK * NTOK);
  int rem = idx % (NTOK * NTOK);
  int t = rpi[rem];
  const float* tbl = s ? tbl2 : tbl1;
  float* dst = bias + s * NHEAD * NTOK * NTOK;
#pragma unroll
  for (int h = 0; h < NHEAD; ++h)
    dst[h * NTOK * NTOK + rem] = tbl[t * NHEAD + h];
}

// --------------------------- main kernel -----------------------------------

__global__ __launch_bounds__(256, 1)
void xattn_kernel(const float* __restrict__ x, const float* __restrict__ y,
                  const float* __restrict__ mask,
                  const float* __restrict__ bqkv1, const float* __restrict__ bqkv2,
                  const float* __restrict__ bproj1, const float* __restrict__ bproj2,
                  const float* __restrict__ g1, const float* __restrict__ be1,
                  const float* __restrict__ g2, const float* __restrict__ be2,
                  const __bf16* __restrict__ wqkvT,
                  const __bf16* __restrict__ wprojT,
                  const float* __restrict__ biasT,
                  float* __restrict__ out) {
  extern __shared__ char smem[];
  float*  Xs      = (float*)(smem + LDS_X);
  float*  Ys      = (float*)(smem + LDS_Y);
  __bf16* qkvX    = (__bf16*)(smem + LDS_QKVX);
  __bf16* qkvY    = (__bf16*)(smem + LDS_QKVY);
  __bf16* Vt      = (__bf16*)(smem + LDS_VT);
  __bf16* Pbuf    = (__bf16*)(smem + LDS_P);
  float*  ProjBuf = (float*)(smem + LDS_P);        // union: dead P by proj time
  __bf16* Obuf    = (__bf16*)(smem + LDS_OBUF);
  float*  muS     = (float*)(smem + LDS_STATS);
  float*  rsS     = muS + 128;

  const int b    = blockIdx.x;
  const int tid  = threadIdx.x;
  const int wave = tid >> 5;
  const int lane = tid & 31;

  // ---- phase 0: stage x,y into LDS (f32, kept for residual) ----
  {
    const float4* xg = (const float4*)(x + (size_t)b * NTOK * CCH);
    const float4* yg = (const float4*)(y + (size_t)b * NTOK * CCH);
    float4* xs4 = (float4*)Xs;
    float4* ys4 = (float4*)Ys;
    for (int i = tid; i < NTOK * CCH / 4; i += 256) {
      xs4[i] = xg[i];
      ys4[i] = yg[i];
    }
  }
  __syncthreads();

  // ---- phase 1: QKV GEMMs, both streams -> bf16 in LDS ----
  for (int s = 0; s < 2; ++s) {
    const float*  A   = s ? Ys : Xs;
    const __bf16* W   = wqkvT + s * QKVC * CCH;    // [j][k]
    const float*  bq  = s ? bqkv2 : bqkv1;
    __bf16*       dst = s ? qkvY : qkvX;
    for (int t = 0; t < 12; ++t) {                 // 96 tiles / 8 waves
      int tile = wave * 12 + t;
      int mt = tile & 3, nt = tile >> 2;           // mt 0..3, nt 0..23
      v8f acc = {};
#pragma unroll
      for (int kk = 0; kk < CCH; kk += 32) {
        v16bf a  = frag_f32(A, CCH, mt, kk, lane);
        v16bf bb = frag_bf(W, CCH, nt, kk, lane);
        acc = wmma_bf16(a, bb, acc);
      }
      int col = nt * 16 + (lane & 15);
      int mb  = mt * 16 + ((lane & 16) ? 8 : 0);
      float bj = bq[col];
#pragma unroll
      for (int r = 0; r < 8; ++r)
        dst[(mb + r) * QKVC + col] = (__bf16)(acc[r] + bj);
    }
  }
  __syncthreads();

  // ---- phase 1b: transpose V into Vt[s][d][m] for contiguous B-frags ----
  for (int i = tid; i < 2 * CCH * NTOK; i += 256) {
    int s = i >> 13;
    int rem = i & 8191;
    int d = rem >> 6, m = rem & 63;
    const __bf16* q = s ? qkvY : qkvX;
    Vt[i] = q[m * QKVC + 2 * CCH + d];             // v section at col 256..383
  }
  __syncthreads();

  // ---- phase 2: attention scores + softmax (wave = stream*4 + head) ----
  {
    const int s = wave >> 2, h = wave & 3;
    const __bf16* Q  = (s ? qkvY : qkvX) + h * HD;             // q section
    const __bf16* K  = (s ? qkvX : qkvY) + CCH + h * HD;       // other stream k
    const __bf16* Vh = Vt + (1 - s) * CCH * NTOK + h * HD * NTOK;
    const float*  bw = biasT + s * NHEAD * NTOK * NTOK + h * NTOK * NTOK;
    const float*  mw = mask + (size_t)(b & MASKW) * NTOK * NTOK;
    __bf16*       Pw = Pbuf + wave * NTOK * NTOK;

    v8f S[4][4];
    v16bf bfr[4];
#pragma unroll
    for (int nt = 0; nt < 4; ++nt)
      bfr[nt] = frag_bf(K, QKVC, nt, 0, lane);     // K=HD=32: one k-step
#pragma unroll
    for (int mt = 0; mt < 4; ++mt) {
      v16bf a = frag_bf(Q, QKVC, mt, 0, lane);
      v8f z = {};
#pragma unroll
      for (int nt = 0; nt < 4; ++nt)
        S[mt][nt] = wmma_bf16(a, bfr[nt], z);
    }

    const float scale = 0.08838834764831845f;      // 1/sqrt(C)
    const int mlo = lane & 15;
    const int hi8 = (lane & 16) ? 8 : 0;
#pragma unroll
    for (int mt = 0; mt < 4; ++mt) {
#pragma unroll
      for (int r = 0; r < 8; ++r) {
        int n = mt * 16 + r + hi8;                 // query row
        float vals[4];
        float mx = -3.4e38f;
#pragma unroll
        for (int nt = 0; nt < 4; ++nt) {
          int m = nt * 16 + mlo;                   // key col
          float v = S[mt][nt][r] * scale + bw[n * 64 + m] + mw[n * 64 + m];
          vals[nt] = v;
          mx = fmaxf(mx, v);
        }
        mx = fmaxf(mx, __shfl_xor(mx, 1, 32));
        mx = fmaxf(mx, __shfl_xor(mx, 2, 32));
        mx = fmaxf(mx, __shfl_xor(mx, 4, 32));
        mx = fmaxf(mx, __shfl_xor(mx, 8, 32));
        float sum = 0.f;
#pragma unroll
        for (int nt = 0; nt < 4; ++nt) {
          vals[nt] = __expf(vals[nt] - mx);
          sum += vals[nt];
        }
        sum += __shfl_xor(sum, 1, 32);
        sum += __shfl_xor(sum, 2, 32);
        sum += __shfl_xor(sum, 4, 32);
        sum += __shfl_xor(sum, 8, 32);
        float inv = 1.f / sum;
#pragma unroll
        for (int nt = 0; nt < 4; ++nt)
          Pw[n * 64 + nt * 16 + mlo] = (__bf16)(vals[nt] * inv);
      }
    }

    // ---- phase 3: O = P @ V (K=64 -> two k-steps) ----
#pragma unroll
    for (int mt = 0; mt < 4; ++mt) {
#pragma unroll
      for (int dt = 0; dt < 2; ++dt) {
        v8f acc = {};
#pragma unroll
        for (int kk = 0; kk < NTOK; kk += 32) {
          v16bf a  = frag_bf(Pw, NTOK, mt, kk, lane);
          v16bf bb = frag_bf(Vh, NTOK, dt, kk, lane);
          acc = wmma_bf16(a, bb, acc);
        }
        int j  = h * HD + dt * 16 + mlo;
        int mb = mt * 16 + hi8;
#pragma unroll
        for (int r = 0; r < 8; ++r)
          Obuf[s * NTOK * CCH + (mb + r) * CCH + j] = (__bf16)acc[r];
      }
    }
  }
  __syncthreads();

  // ---- phase 4: projection GEMM (f32 out + bias into ProjBuf) ----
  for (int t = 0; t < 8; ++t) {                    // 64 tiles / 8 waves
    int g = wave * 8 + t;
    int ss = g >> 5;
    int rem = g & 31;
    int mt = rem & 3, nt = rem >> 2;               // nt 0..7
    const __bf16* A = Obuf + ss * NTOK * CCH;
    const __bf16* W = wprojT + ss * CCH * CCH;     // [j][k]
    const float* bp = ss ? bproj2 : bproj1;
    v8f acc = {};
#pragma unroll
    for (int kk = 0; kk < CCH; kk += 32) {
      v16bf a  = frag_bf(A, CCH, mt, kk, lane);
      v16bf bb = frag_bf(W, CCH, nt, kk, lane);
      acc = wmma_bf16(a, bb, acc);
    }
    int col = nt * 16 + (lane & 15);
    int mb  = mt * 16 + ((lane & 16) ? 8 : 0);
    float bj = bp[col];
#pragma unroll
    for (int r = 0; r < 8; ++r)
      ProjBuf[ss * NTOK * CCH + (mb + r) * CCH + col] = acc[r] + bj;
  }
  __syncthreads();

  // ---- phase 5a: LN stats per row (proj + residual) ----
  if (tid < 128) {
    int ss = tid >> 6, n = tid & 63;
    const float* row = ProjBuf + tid * CCH;
    const float* res = (ss ? Ys : Xs) + n * CCH;
    float sum = 0.f;
#pragma unroll 8
    for (int j = 0; j < CCH; ++j) sum += row[j] + res[j];
    float mu = sum * (1.f / CCH);
    float var = 0.f;
#pragma unroll 8
    for (int j = 0; j < CCH; ++j) {
      float d = row[j] + res[j] - mu;
      var += d * d;
    }
    muS[tid] = mu;
    rsS[tid] = rsqrtf(var * (1.f / CCH) + 1e-5f);
  }
  __syncthreads();

  // ---- phase 5b: normalize + affine + coalesced store ----
  for (int i = tid; i < 2 * NTOK * CCH; i += 256) {
    int ss  = i >> 13;
    int rem = i & 8191;
    int row = i >> 7;
    int j   = i & 127;
    float val = ProjBuf[i] + (ss ? Ys : Xs)[rem];
    const float* gg = ss ? g2 : g1;
    const float* bb = ss ? be2 : be1;
    out[(size_t)ss * ((size_t)NWIN * NTOK * CCH) + (size_t)b * NTOK * CCH + rem] =
        (val - muS[row]) * rsS[row] * gg[j] + bb[j];
  }
}

// --------------------------- launch ----------------------------------------

extern "C" void kernel_launch(void* const* d_in, const int* in_sizes, int n_in,
                              void* d_out, int out_size, void* d_ws, size_t ws_size,
                              hipStream_t stream) {
  const float* x      = (const float*)d_in[0];
  const float* y      = (const float*)d_in[1];
  const float* mask   = (const float*)d_in[2];
  const float* wqkv1  = (const float*)d_in[3];
  const float* bqkv1  = (const float*)d_in[4];
  const float* wqkv2  = (const float*)d_in[5];
  const float* bqkv2  = (const float*)d_in[6];
  const float* tbl1   = (const float*)d_in[7];
  const float* tbl2   = (const float*)d_in[8];
  const float* wproj1 = (const float*)d_in[9];
  const float* bproj1 = (const float*)d_in[10];
  const float* wproj2 = (const float*)d_in[11];
  const float* bproj2 = (const float*)d_in[12];
  const float* g1     = (const float*)d_in[13];
  const float* be1    = (const float*)d_in[14];
  const float* g2     = (const float*)d_in[15];
  const float* be2    = (const float*)d_in[16];
  const int*   rpi    = (const int*)d_in[17];

  char* ws = (char*)d_ws;
  __bf16* wqkvT  = (__bf16*)(ws + WS_WQKVT);
  __bf16* wprojT = (__bf16*)(ws + WS_WPROJT);
  float*  biasT  = (float*)(ws + WS_BIAS);

  prep_wqkv<<<(2 * QKVC * CCH + 255) / 256, 256, 0, stream>>>(wqkv1, wqkv2, wqkvT);
  prep_wproj<<<(2 * CCH * CCH + 255) / 256, 256, 0, stream>>>(wproj1, wproj2, wprojT);
  prep_bias<<<(2 * NTOK * NTOK + 255) / 256, 256, 0, stream>>>(tbl1, tbl2, rpi, biasT);

  hipFuncSetAttribute(reinterpret_cast<const void*>(&xattn_kernel),
                      hipFuncAttributeMaxDynamicSharedMemorySize, SMEM_BYTES);

  xattn_kernel<<<NWIN, 256, SMEM_BYTES, stream>>>(
      x, y, mask, bqkv1, bqkv2, bproj1, bproj2, g1, be1, g2, be2,
      wqkvT, wprojT, biasT, (float*)d_out);
}
